// METALayer_32444182954028
// MI455X (gfx1250) — compile-verified, matrix operand order checked by reading, not run on hardware
//
#include <hip/hip_runtime.h>

// ============================================================================
// META hierarchical-attention forward for MI455X (gfx1250, wave32, WMMA).
// Dense projections: v_wmma_f32_16x16x32_bf16. Each wave computes a 16x64
// output strip (4 n-tiles) reusing one A fragment per k-step. All fragment
// loads for a k-step are staged before the WMMA chain so the scheduler can
// retire WMMAs against partial load counts. B pre-packed into exact CDNA5
// fragment layout. Edge attention (scatter softmax) fp32, L2-resident gathers.
// ============================================================================

#define DEV __device__ __forceinline__

typedef __attribute__((ext_vector_type(16))) __bf16 v16bf;
typedef __attribute__((ext_vector_type(8)))  float  v8f;

union FragB { v16bf v; unsigned short s[16]; unsigned int u[8]; uint4 q[2]; };

DEV unsigned short f2b(float x) {           // fp32 -> bf16 RNE
  unsigned int u = __float_as_uint(x);
  u += 0x7FFFu + ((u >> 16) & 1u);
  return (unsigned short)(u >> 16);
}
// order-preserving float <-> uint transform for atomic segment-max
DEV unsigned int encf(float x) {
  unsigned int u = __float_as_uint(x);
  return (u & 0x80000000u) ? ~u : (u | 0x80000000u);
}
DEV float decf(unsigned int k) {
  unsigned int u = (k & 0x80000000u) ? (k ^ 0x80000000u) : ~k;
  return __uint_as_float(u);
}

// ---------------------------------------------------------------------------
// LayerNorm (D=128) -> bf16 ; one wave32 per row, 4 elems/lane
// ---------------------------------------------------------------------------
__global__ void ln_bf16(const float* __restrict__ X, const float* __restrict__ g,
                        const float* __restrict__ b, unsigned short* __restrict__ Y,
                        int M) {
  int row  = blockIdx.x * 8 + (threadIdx.x >> 5);
  int lane = threadIdx.x & 31;
  if (row >= M) return;
  float4 x = ((const float4*)(X + (size_t)row * 128))[lane];
  float s = x.x + x.y + x.z + x.w;
#pragma unroll
  for (int o = 16; o > 0; o >>= 1) s += __shfl_xor(s, o, 32);
  float mean = s * (1.0f / 128.0f);
  float dx = x.x - mean, dy = x.y - mean, dz = x.z - mean, dw = x.w - mean;
  float v = dx * dx + dy * dy + dz * dz + dw * dw;
#pragma unroll
  for (int o = 16; o > 0; o >>= 1) v += __shfl_xor(v, o, 32);
  float inv = rsqrtf(v * (1.0f / 128.0f) + 1e-5f);
  float4 gg = ((const float4*)g)[lane];
  float4 bb = ((const float4*)b)[lane];
  unsigned int lo = (unsigned int)f2b(dx * inv * gg.x + bb.x) |
                    ((unsigned int)f2b(dy * inv * gg.y + bb.y) << 16);
  unsigned int hi = (unsigned int)f2b(dz * inv * gg.z + bb.z) |
                    ((unsigned int)f2b(dw * inv * gg.w + bb.w) << 16);
  ((uint2*)(Y + (size_t)row * 128))[lane] = make_uint2(lo, hi);
}

// ---------------------------------------------------------------------------
// Pack fp32 weight W[K x Nc] into per-lane WMMA B-fragment order (bf16):
// fragment (nt, ks, lane) = 8 u32; u32 v2 covers k = koff(v2)+lh*8+ks*32,
// koff = {0,2,4,6,16,18,20,22}, col = nt*16 + (lane&15) (clamped).
// ---------------------------------------------------------------------------
__global__ void pack_w_bf16(const float* __restrict__ W, unsigned int* __restrict__ P,
                            int Nc, int ksteps, int total) {
  int i = blockIdx.x * blockDim.x + threadIdx.x;
  if (i >= total) return;
  int v2   = i & 7;
  int lane = (i >> 3) & 31;
  int frag = i >> 8;                   // nt * ksteps + ks
  int ks   = frag % ksteps;
  int nt   = frag / ksteps;
  int lr = lane & 15, lh = lane >> 4;
  int col = nt * 16 + lr; if (col >= Nc) col = Nc - 1;
  int k = ((v2 < 4) ? (v2 * 2) : (16 + (v2 - 4) * 2)) + lh * 8 + ks * 32;
  unsigned short lo = f2b(W[(size_t)k * Nc + col]);
  unsigned short hi = f2b(W[(size_t)(k + 1) * Nc + col]);
  P[i] = (unsigned int)lo | ((unsigned int)hi << 16);
}

// ---------------------------------------------------------------------------
// WMMA GEMM: C[M x Nc] = act( A_bf16[M x K] @ W_bf16[K x Nc] + bias )
//   ACT : 0 none, 1 sigmoid, 2 exact GELU
//   OUTM: 0 store f32, 1 accumulate into f32 (residual), 2 store bf16
//   NT  : n-tiles (16 cols each) per wave -> A fragment reused NT times
// One 16x(16*NT) strip per wave32; 8 waves/block = 128 rows.
// ---------------------------------------------------------------------------
template <int ACT, int OUTM, int K, int NT>
__global__ void gemm_wmma(const unsigned short* __restrict__ A,
                          const uint4* __restrict__ Wp,
                          const float* __restrict__ bias,
                          float* __restrict__ Cf, unsigned short* __restrict__ Cb,
                          int M, int Nc) {
  constexpr int KSTEPS = K / 32;
  int wave = threadIdx.x >> 5, lane = threadIdx.x & 31;
  int m0 = blockIdx.x * 128 + wave * 16;
  if (m0 >= M) return;                    // whole-wave uniform exit
  int n0 = blockIdx.y * (16 * NT);
  int lr = lane & 15, lh = lane >> 4;
  int row = m0 + lr; if (row >= M) row = M - 1;   // clamp: loads unconditional
  const unsigned short* arow = A + (size_t)row * K + lh * 8;
  const uint4* bp = Wp + ((size_t)blockIdx.y * NT * KSTEPS * 32 + lane) * 2;

  v8f acc[NT];
#pragma unroll
  for (int t = 0; t < NT; ++t) acc[t] = (v8f){};
#pragma unroll
  for (int ks = 0; ks < KSTEPS; ++ks) {
    // stage ALL fragment loads for this k-step first (one clause batch),
    // then chain WMMAs so they can retire on partial load counts
    FragB a, b[NT];
    const uint4* ap = (const uint4*)(arow + ks * 32);
    a.q[0] = ap[0];                       // elements 0..7   (of this k-half)
    a.q[1] = ap[2];                       // elements 16..23 (+32B)
#pragma unroll
    for (int t = 0; t < NT; ++t) {
      b[t].q[0] = bp[(size_t)(t * KSTEPS + ks) * 64 + 0];
      b[t].q[1] = bp[(size_t)(t * KSTEPS + ks) * 64 + 1];
    }
#pragma unroll
    for (int t = 0; t < NT; ++t)
      acc[t] = __builtin_amdgcn_wmma_f32_16x16x32_bf16(false, a.v, false, b[t].v,
                                                       (short)0, acc[t], false, false);
  }
#pragma unroll
  for (int t = 0; t < NT; ++t) {
#pragma unroll
    for (int j = 0; j < 8; ++j) {         // C/D: VGPR j -> M = j (+8 hi lanes)
      int m = m0 + j + lh * 8;
      int n = n0 + t * 16 + lr;
      if (m < M && n < Nc) {
        float val = acc[t][j];
        if (bias) val += bias[n];
        if (ACT == 1) val = 1.0f / (1.0f + expf(-val));
        if (ACT == 2) val = 0.5f * val * (1.0f + erff(val * 0.7071067811865475f));
        size_t o = (size_t)m * Nc + n;
        if (OUTM == 0) Cf[o] = val;
        else if (OUTM == 1) Cf[o] += val;
        else Cb[o] = f2b(val);
      }
    }
  }
}

// ---------------------------------------------------------------------------
// Edge pass 1: scores[e,h] = dot16(Q[src],K[dst])*0.25 + bsrc[src,h] (+bdst[dst,h])
// ---------------------------------------------------------------------------
__global__ void edge_score(const int* __restrict__ src, const int* __restrict__ dst,
                           int E, const float* __restrict__ Q, int ldq,
                           const float* __restrict__ K, int ldk,
                           const float* __restrict__ bsrc, const float* __restrict__ bdst,
                           float* __restrict__ scores, unsigned int* __restrict__ segmax) {
  int t = blockIdx.x * blockDim.x + threadIdx.x;
  if (t >= E * 8) return;
  int e = t >> 3, h = t & 7;
  int s = src[e], d = dst[e];
  const float4* q = (const float4*)(Q + (size_t)s * ldq + h * 16);
  const float4* k = (const float4*)(K + (size_t)d * ldk + h * 16);
  float acc = 0.0f;
#pragma unroll
  for (int i = 0; i < 4; ++i) {
    float4 a = q[i], c = k[i];
    acc += a.x * c.x + a.y * c.y + a.z * c.z + a.w * c.w;
  }
  acc = acc * 0.25f + bsrc[s * 8 + h];
  if (bdst) acc += bdst[d * 8 + h];
  scores[t] = acc;
  atomicMax(segmax + s * 8 + h, encf(acc));
}

// Edge pass 2: e = exp(score - segmax) ; segment sum
__global__ void edge_exp(const int* __restrict__ src, int E,
                         float* __restrict__ scores,
                         const unsigned int* __restrict__ segmax,
                         float* __restrict__ segsum) {
  int t = blockIdx.x * blockDim.x + threadIdx.x;
  if (t >= E * 8) return;
  int e = t >> 3, h = t & 7;
  int s = src[e];
  float ex = expf(scores[t] - decf(segmax[s * 8 + h]));
  scores[t] = ex;
  atomicAdd(segsum + s * 8 + h, ex);
}

// Edge pass 3: out[src] += (e/sum) * (gs[dst]?) * V[dst]
__global__ void edge_acc(const int* __restrict__ src, const int* __restrict__ dst,
                         int E, const float* __restrict__ scores,
                         const float* __restrict__ segsum,
                         const float* __restrict__ V, int ldv,
                         const float* __restrict__ gs, float* __restrict__ out) {
  int t = blockIdx.x * blockDim.x + threadIdx.x;
  if (t >= E * 8) return;
  int e = t >> 3, h = t & 7;
  int s = src[e], d = dst[e];
  float w = scores[t] / (segsum[s * 8 + h] + 1e-12f);
  const float4* vp = (const float4*)(V + (size_t)d * ldv + h * 16);
  const float4* gp = gs ? (const float4*)(gs + (size_t)d * 128 + h * 16) : nullptr;
  float* op = out + (size_t)s * 128 + h * 16;
#pragma unroll
  for (int i = 0; i < 4; ++i) {
    float4 v = vp[i];
    if (gp) { float4 g = gp[i]; v.x *= g.x; v.y *= g.y; v.z *= g.z; v.w *= g.w; }
    atomicAdd(op + i * 4 + 0, w * v.x);
    atomicAdd(op + i * 4 + 1, w * v.y);
    atomicAdd(op + i * 4 + 2, w * v.z);
    atomicAdd(op + i * 4 + 3, w * v.w);
  }
}

// gated output -> bf16 for the wo projection
__global__ void gate_mul_bf16(const float* __restrict__ gate, const float* __restrict__ a,
                              unsigned short* __restrict__ o, int n) {
  int i = blockIdx.x * blockDim.x + threadIdx.x;
  if (i < n) o[i] = f2b(gate[i] * a[i]);
}

// ===========================================================================
extern "C" void kernel_launch(void* const* d_in, const int* in_sizes, int n_in,
                              void* d_out, int out_size, void* d_ws, size_t ws_size,
                              hipStream_t stream) {
  (void)n_in; (void)out_size; (void)ws_size;
  // ---- input map -----------------------------------------------------------
  // 0..3 h0..h3 ; 4..11 nbr{r}_src/dst ; 12..17 incidence ; 18.. params
  const int NC[4] = { in_sizes[0] / 128, in_sizes[1] / 128,
                      in_sizes[2] / 128, in_sizes[3] / 128 };
  const int NMAX = 80000;
  const long EMAX = 800000;

  auto pf = [&](int i) { return (const float*)d_in[i]; };
  auto pi = [&](int i) { return (const int*)d_in[i]; };

  // params flattened in jax pytree order (dict keys sorted alphabetically)
  const int P_DN = 18, P_FFN = 45, P_INTRA = 61;
  const int P_LN_DN_S = 81, P_LN_DN_T = 87, P_LN_FFN = 93, P_LN_INTRA = 101;
  const int P_LN_UP_S = 109, P_LN_UP_T = 115, P_UP = 121;

  // ---- workspace carve -----------------------------------------------------
  char* wsp = (char*)d_ws; size_t off = 0;
  auto carve = [&](size_t bytes) -> void* {
    void* p = wsp + off; off = (off + bytes + 255) & ~(size_t)255; return p;
  };
  unsigned short* us_Aln = (unsigned short*)carve((size_t)NMAX * 128 * 2);
  unsigned short* us_Bln = (unsigned short*)carve((size_t)NMAX * 128 * 2);
  float* f_QKV  = (float*)carve((size_t)NMAX * 384 * 4);   // intra QKV / cross Q+KV
  float* f_KV   = f_QKV + (size_t)NMAX * 128;              // cross KV region
  float* f_gate = (float*)carve((size_t)NMAX * 128 * 4);   // gate / gt
  float* f_gs   = (float*)carve((size_t)NMAX * 128 * 4);   // cross source gate
  float* f_eb   = (float*)carve((size_t)NMAX * 8 * 4);     // wb / wbt proj
  float* f_bs   = (float*)carve((size_t)NMAX * 8 * 4);     // wbs proj
  float* f_scores = (float*)carve((size_t)EMAX * 8 * 4);
  unsigned int* u_segmax = (unsigned int*)carve((size_t)NMAX * 8 * 4);
  float* f_segsum = (float*)carve((size_t)NMAX * 8 * 4);
  float* f_attn = (float*)carve((size_t)NMAX * 128 * 4);
  unsigned short* us_tmp = (unsigned short*)carve((size_t)NMAX * 512 * 2);
  unsigned int* us_wp = (unsigned int*)carve((size_t)128 * 512 * 4); // packed W

  // ---- output (residual stream) --------------------------------------------
  float* ht[4];
  ht[0] = (float*)d_out;
  for (int r = 1; r < 4; ++r) ht[r] = ht[r - 1] + (size_t)NC[r - 1] * 128;
  for (int r = 0; r < 4; ++r)
    hipMemcpyAsync(ht[r], d_in[r], (size_t)NC[r] * 128 * sizeof(float),
                   hipMemcpyDeviceToDevice, stream);

  // ---- helpers -------------------------------------------------------------
  auto gemm = [&](const unsigned short* A, int K, const float* W32,
                  const float* bias, float* Cf, unsigned short* Cb,
                  int M, int Nc, int act, int outm) {
    int ntiles = (Nc + 15) / 16, ksteps = K >> 5;
    int total = ntiles * ksteps * 256;    // fragments * 32 lanes * 8 u32
    pack_w_bf16<<<(total + 255) / 256, 256, 0, stream>>>(W32, us_wp, Nc, ksteps, total);
    const uint4* Wp = (const uint4*)us_wp;
    if (Nc % 64 == 0) {                   // NT = 4 (all large GEMMs)
      dim3 g((M + 127) / 128, ntiles / 4);
      if (K == 512)
        gemm_wmma<0, 1, 512, 4><<<g, 256, 0, stream>>>(A, Wp, bias, Cf, Cb, M, Nc);
      else if (act == 0 && outm == 0)
        gemm_wmma<0, 0, 128, 4><<<g, 256, 0, stream>>>(A, Wp, bias, Cf, Cb, M, Nc);
      else if (act == 1 && outm == 0)
        gemm_wmma<1, 0, 128, 4><<<g, 256, 0, stream>>>(A, Wp, bias, Cf, Cb, M, Nc);
      else if (act == 0 && outm == 1)
        gemm_wmma<0, 1, 128, 4><<<g, 256, 0, stream>>>(A, Wp, bias, Cf, Cb, M, Nc);
      else
        gemm_wmma<2, 2, 128, 4><<<g, 256, 0, stream>>>(A, Wp, bias, Cf, Cb, M, Nc);
    } else {                              // NT = 1 (Nc == 8 bias projections)
      dim3 g((M + 127) / 128, ntiles);
      gemm_wmma<0, 0, 128, 1><<<g, 256, 0, stream>>>(A, Wp, bias, Cf, Cb, M, Nc);
    }
  };
  auto ln = [&](const float* X, const float* g, const float* b,
                unsigned short* Y, int M) {
    ln_bf16<<<(M + 7) / 8, 256, 0, stream>>>(X, g, b, Y, M);
  };
  auto edges = [&](const int* src, const int* dst, int E, int Nseg,
                   const float* Q, int ldq, const float* K, int ldk,
                   const float* bsrc, const float* bdst,
                   const float* V, int ldv, const float* gs) {
    hipMemsetAsync(u_segmax, 0, (size_t)Nseg * 8 * 4, stream);
    hipMemsetAsync(f_segsum, 0, (size_t)Nseg * 8 * 4, stream);
    hipMemsetAsync(f_attn, 0, (size_t)Nseg * 128 * 4, stream);
    int T = E * 8, B = (T + 255) / 256;
    edge_score<<<B, 256, 0, stream>>>(src, dst, E, Q, ldq, K, ldk, bsrc, bdst,
                                      f_scores, u_segmax);
    edge_exp<<<B, 256, 0, stream>>>(src, E, f_scores, u_segmax, f_segsum);
    edge_acc<<<B, 256, 0, stream>>>(src, dst, E, f_scores, f_segsum, V, ldv,
                                    gs, f_attn);
  };

  // ---- 1) intra (self) attention per level ---------------------------------
  for (int r = 0; r < 4; ++r) {
    int Nr = NC[r];
    const int* src = pi(4 + 2 * r);
    const int* dst = pi(5 + 2 * r);
    int E = in_sizes[4 + 2 * r];
    int o = P_INTRA + r * 5;   // gate_b, gate_w, qkv, wb, wo
    const float *gate_b = pf(o + 0), *gate_w = pf(o + 1), *qkv = pf(o + 2),
                *wb = pf(o + 3), *wo = pf(o + 4);
    const float *lnb = pf(P_LN_INTRA + r * 2 + 0), *lng = pf(P_LN_INTRA + r * 2 + 1);

    ln(ht[r], lng, lnb, us_Aln, Nr);
    gemm(us_Aln, 128, qkv, nullptr, f_QKV, nullptr, Nr, 384, 0, 0);
    gemm(us_Aln, 128, gate_w, gate_b, f_gate, nullptr, Nr, 128, 1, 0);
    gemm(us_Aln, 128, wb, nullptr, f_eb, nullptr, Nr, 8, 0, 0);
    edges(src, dst, E, Nr, f_QKV, 384, f_QKV + 128, 384, f_eb, nullptr,
          f_QKV + 256, 384, nullptr);
    int n = Nr * 128;
    gate_mul_bf16<<<(n + 255) / 256, 256, 0, stream>>>(f_gate, f_attn, us_tmp, n);
    gemm(us_tmp, 128, wo, nullptr, ht[r], nullptr, Nr, 128, 0, 1);
  }

  // ---- 2) cross attentions (up then down) ----------------------------------
  auto cross = [&](int pbase, int lnTbase, int lnSbase, int i,
                   const int* it, const int* isr, int E, int tr, int sr) {
    int o = pbase + i * 9;  // gs_b, gs_w, gt_b, gt_w, wbs, wbt, wkv, wo, wq
    const float *gs_b = pf(o + 0), *gs_w = pf(o + 1), *gt_b = pf(o + 2),
                *gt_w = pf(o + 3), *wbs = pf(o + 4), *wbt = pf(o + 5),
                *wkv = pf(o + 6), *wo = pf(o + 7), *wq = pf(o + 8);
    const float *lnTb = pf(lnTbase + i * 2 + 0), *lnTg = pf(lnTbase + i * 2 + 1);
    const float *lnSb = pf(lnSbase + i * 2 + 0), *lnSg = pf(lnSbase + i * 2 + 1);
    int Nt = NC[tr], Ns = NC[sr];

    ln(ht[tr], lnTg, lnTb, us_Aln, Nt);
    ln(ht[sr], lnSg, lnSb, us_Bln, Ns);
    gemm(us_Aln, 128, wq, nullptr, f_QKV, nullptr, Nt, 128, 0, 0);
    gemm(us_Bln, 128, wkv, nullptr, f_KV, nullptr, Ns, 256, 0, 0);
    gemm(us_Aln, 128, wbt, nullptr, f_eb, nullptr, Nt, 8, 0, 0);
    gemm(us_Bln, 128, wbs, nullptr, f_bs, nullptr, Ns, 8, 0, 0);
    gemm(us_Aln, 128, gt_w, gt_b, f_gate, nullptr, Nt, 128, 1, 0);
    gemm(us_Bln, 128, gs_w, gs_b, f_gs, nullptr, Ns, 128, 1, 0);
    edges(it, isr, E, Nt, f_QKV, 128, f_KV, 256, f_eb, f_bs,
          f_KV + 128, 256, f_gs);
    int n = Nt * 128;
    gate_mul_bf16<<<(n + 255) / 256, 256, 0, stream>>>(f_gate, f_attn, us_tmp, n);
    gemm(us_tmp, 128, wo, nullptr, ht[tr], nullptr, Nt, 128, 0, 1);
  };

  { // up: (inc_01_edge, inc_01_node, 1), (inc_12_bend, inc_12_edge, 2), (inc_23_torsion, inc_23_bend, 3)
    const int itI[3] = {12, 14, 16}, isI[3] = {13, 15, 17}, trU[3] = {1, 2, 3};
    for (int i = 0; i < 3; ++i)
      cross(P_UP, P_LN_UP_T, P_LN_UP_S, i, pi(itI[i]), pi(isI[i]),
            in_sizes[itI[i]], trU[i], trU[i] - 1);
  }
  { // dn: (inc_23_bend, inc_23_torsion, 2), (inc_12_edge, inc_12_bend, 1), (inc_01_node, inc_01_edge, 0)
    const int itI[3] = {17, 15, 13}, isI[3] = {16, 14, 12}, trD[3] = {2, 1, 0};
    for (int i = 0; i < 3; ++i)
      cross(P_DN, P_LN_DN_T, P_LN_DN_S, i, pi(itI[i]), pi(isI[i]),
            in_sizes[itI[i]], trD[i], trD[i] + 1);
  }

  // ---- 3) FFN per level ----------------------------------------------------
  for (int r = 0; r < 4; ++r) {
    int Nr = NC[r];
    int o = P_FFN + r * 4;  // b1, b2, w1, w2
    const float *b1 = pf(o + 0), *b2 = pf(o + 1), *w1 = pf(o + 2), *w2 = pf(o + 3);
    const float *lnb = pf(P_LN_FFN + r * 2 + 0), *lng = pf(P_LN_FFN + r * 2 + 1);
    ln(ht[r], lng, lnb, us_Aln, Nr);
    gemm(us_Aln, 128, w1, b1, nullptr, us_tmp, Nr, 512, 2, 2);   // GELU -> bf16
    gemm(us_tmp, 512, w2, b2, ht[r], nullptr, Nr, 128, 0, 1);    // += residual
  }
}
